// CGNLBlock_17411797418308
// MI455X (gfx1250) — matrix operand order repaired and checked by reading, not compiled, not run on hardware
//
#include <hip/hip_runtime.h>
#include <hip/hip_bf16.h>

// CGNL non-local block: B=4, C=512, I=256, N=4096, f32 in/out.
// f16 WMMA everywhere; flash-softmax fusion; TDM double-buffered K/V tiles.

#define BB 4
#define CC 512
#define II 256
#define NN 4096

typedef _Float16 h16;
typedef __attribute__((ext_vector_type(8)))  _Float16 v8h;
typedef __attribute__((ext_vector_type(16))) _Float16 v16h;
typedef __attribute__((ext_vector_type(8)))  float    v8f;
typedef __attribute__((ext_vector_type(4)))  unsigned int v4u;
typedef __attribute__((ext_vector_type(4)))  int v4i;
typedef __attribute__((ext_vector_type(8)))  int v8i;

__device__ __forceinline__ v8f wmma_f16(v16h a, v16h b, v8f c) {
  // D = A(16x32 f16) * B(32x16 f16) + C(16x16 f32)
  return __builtin_amdgcn_wmma_f32_16x16x32_f16(false, a, false, b, (short)0, c,
                                                false, false);
}

// A-fragment (16x32 f16): lane holds row = lane%16; elements 0..15 = K
// [hf*16, hf*16+16)  (hf = lane/16).
__device__ __forceinline__ v16h load_a_frag(const h16* rowp, int hf) {
  v8h lo = *(const v8h*)(rowp + hf * 16);
  v8h hi = *(const v8h*)(rowp + hf * 16 + 8);
  return __builtin_shufflevector(lo, hi, 0, 1, 2, 3, 4, 5, 6, 7, 8, 9, 10, 11,
                                 12, 13, 14, 15);
}

// Tensor Data Mover: 2D tile (x in 8-byte units, contiguous; y rows with
// stride_x8 8-byte units) from global -> LDS.  D# per CDNA5 ISA ch.8:
// group0 = {count, lds_addr, global_addr, type=2}, group1 = dims/strides.
// amdgpu-toolchain (clang-23) 6-arg builtin form.
__device__ __forceinline__ void tdm_load_2d(unsigned lds_addr, const void* gptr,
                                            unsigned tile_x8, unsigned tile_y,
                                            unsigned stride_x8) {
  unsigned long long ga = (unsigned long long)gptr;
  v4u g0;
  g0[0] = 1u;                                 // count=1, user descriptor
  g0[1] = lds_addr;                           // LDS byte address
  g0[2] = (unsigned)ga;                       // global_addr[31:0]
  g0[3] = (unsigned)(ga >> 32) | (2u << 30);  // global_addr[56:32] | type=2
  v8i g1;
  g1[0] = 3 << 16;                 // data_size = 8B; mask/flags/pad = 0
  g1[1] = (int)(tile_x8 << 16);    // tensor_dim0[15:0] (exact fit, no OOB)
  g1[2] = (int)(tile_y << 16);     // tensor_dim0[31:16]=0, tensor_dim1[15:0]
  g1[3] = (int)(tile_x8 << 16);    // tensor_dim1[31:16]=0, tile_dim0
  g1[4] = (int)tile_y;             // tile_dim1; tile_dim2 = 0
  g1[5] = (int)stride_x8;          // tensor_dim0_stride[31:0]
  g1[6] = 0;                       // stride hi, tensor_dim1_stride lo (unused)
  g1[7] = 0;
  v4i z4 = {0, 0, 0, 0};
  v8i z8 = {0, 0, 0, 0, 0, 0, 0, 0};
  __builtin_amdgcn_tensor_load_to_lds(g0, g1, z4, z4, z8, 0);
}

// ---------------------------------------------------------------------------
// Phase 0: f32 -> f16 weight conversion, and x (B,C,N) -> xt (B,N,C) f16.
// ---------------------------------------------------------------------------
__global__ void cvt_f16_kernel(const float* __restrict__ src,
                               h16* __restrict__ dst, int n) {
  int i = blockIdx.x * blockDim.x + threadIdx.x;
  if (i < n) dst[i] = (h16)src[i];
}

__global__ void transpose_x_kernel(const float* __restrict__ x,
                                   h16* __restrict__ xt) {
  int idx = blockIdx.x * blockDim.x + threadIdx.x;  // flat over B*N*C
  int c = idx & (CC - 1);
  int n = (idx >> 9) & (NN - 1);  // CC = 2^9
  int b = idx >> 21;              // 9 + 12
  xt[idx] = (h16)x[((size_t)b * CC + c) * NN + n];
}

// ---------------------------------------------------------------------------
// Phase 1: theta/phi/g = W (I,C) @ x (C,N) + bias.  One 16x16 tile per wave,
// the three projections share the x B-fragments.
//   thetaT, phiT : (B, N, I) f16     gmat : (B, I, N) f16
// ---------------------------------------------------------------------------
__global__ void __launch_bounds__(128) proj_kernel(
    const h16* __restrict__ xt, const h16* __restrict__ wth,
    const h16* __restrict__ wph, const h16* __restrict__ wg,
    const float* __restrict__ thb, const float* __restrict__ phb,
    const float* __restrict__ gb, h16* __restrict__ thetaT,
    h16* __restrict__ phiT, h16* __restrict__ gmat) {
  const int lane = threadIdx.x & 31, wave = threadIdx.x >> 5;
  const int col = lane & 15, hf = lane >> 4;
  const int n0 = (blockIdx.x * 4 + wave) * 16;
  const int i0 = blockIdx.y * 16;
  const int b = blockIdx.z;

  const h16* xrow = xt + ((size_t)b * NN + n0 + col) * CC + hf * 16;
  const h16* at = wth + (size_t)(i0 + col) * CC;
  const h16* ap = wph + (size_t)(i0 + col) * CC;
  const h16* ag = wg + (size_t)(i0 + col) * CC;

  v8f ct = {}, cp = {}, cg = {};
#pragma unroll
  for (int ks = 0; ks < CC / 32; ++ks) {
    v16h bf = *(const v16h*)(xrow + ks * 32);
    ct = wmma_f16(load_a_frag(at + ks * 32, hf), bf, ct);
    cp = wmma_f16(load_a_frag(ap + ks * 32, hf), bf, cp);
    cg = wmma_f16(load_a_frag(ag + ks * 32, hf), bf, cg);
  }

  v8f bt = *(const v8f*)(thb + i0 + hf * 8);
  v8f bp = *(const v8f*)(phb + i0 + hf * 8);
  v8f bg = *(const v8f*)(gb + i0 + hf * 8);

  v8h st, sp;
#pragma unroll
  for (int v = 0; v < 8; ++v) {
    st[v] = (h16)(ct[v] + bt[v]);
    sp[v] = (h16)(cp[v] + bp[v]);
  }
  size_t rowoff = ((size_t)b * NN + n0 + col) * II + i0 + hf * 8;
  *(v8h*)(thetaT + rowoff) = st;
  *(v8h*)(phiT + rowoff) = sp;
#pragma unroll
  for (int v = 0; v < 8; ++v) {
    gmat[((size_t)b * II + i0 + hf * 8 + v) * NN + n0 + col] =
        (h16)(cg[v] + bg[v]);
  }
}

// ---------------------------------------------------------------------------
// Phase 2: fused flash attention.  Per block: 4 waves x 16 query rows; the
// 32-wide phi/g tiles are TDM-DMA'd into double-buffered LDS once per block
// (4x less L2 traffic than per-wave loads).  Per m-step per wave: 16 WMMAs
// for S, online softmax (stats within 16-lane halves), LDS transpose of P,
// 16 WMMAs for O += P @ g^T.  Output oT: (B, N, I) f16, normalized.
// ---------------------------------------------------------------------------
__global__ void __launch_bounds__(128) attn_kernel(
    const h16* __restrict__ thetaT, const h16* __restrict__ phiT,
    const h16* __restrict__ gmat, h16* __restrict__ oT) {
  __shared__ __align__(128) h16 plds[2][32 * II];    // key tile: [m_local][i]
  __shared__ __align__(128) h16 glds[2][II * 32];    // val tile: [i][m_local]
  __shared__ __align__(32) h16 pstage[4][16 * 32];   // per-wave P transpose

  const int lane = threadIdx.x & 31, wave = threadIdx.x >> 5;
  const int col = lane & 15, hf = lane >> 4;
  const int n0 = (blockIdx.x * 4 + wave) * 16;
  const int b = blockIdx.z;

  // Query A-fragments (read once): A[n][k=i] = thetaT[n][i].
  v16h af[II / 32];
  const h16* arow = thetaT + ((size_t)b * NN + n0 + col) * II;
#pragma unroll
  for (int k = 0; k < II / 32; ++k) af[k] = load_a_frag(arow + k * 32, hf);

  v8f acc[16];
  v8f zero = {};
#pragma unroll
  for (int t = 0; t < 16; ++t) acc[t] = zero;
  float mrow[8], lrow[8];
#pragma unroll
  for (int v = 0; v < 8; ++v) {
    mrow[v] = -1e30f;
    lrow[v] = 0.0f;
  }

  const float SCALE = 0.0625f;  // 1/sqrt(I)

  // Prologue: DMA tile 0 into buffer 0 (wave 0 issues for the whole block).
  if (wave == 0) {
    tdm_load_2d((unsigned)(size_t)&plds[0][0],
                phiT + (size_t)b * NN * II, 64, 32, 64);
    tdm_load_2d((unsigned)(size_t)&glds[0][0],
                gmat + (size_t)b * II * NN, 8, 256, 1024);
  }

  int buf = 0;
  for (int ms = 0; ms < NN; ms += 32) {
    if (wave == 0) __builtin_amdgcn_s_wait_tensorcnt(0);
    __syncthreads();  // tiles[buf] ready; prior readers of buf^1 all done
    if (wave == 0 && ms + 32 < NN) {
      tdm_load_2d((unsigned)(size_t)&plds[buf ^ 1][0],
                  phiT + ((size_t)b * NN + ms + 32) * II, 64, 32, 64);
      tdm_load_2d((unsigned)(size_t)&glds[buf ^ 1][0],
                  gmat + (size_t)b * II * NN + ms + 32, 8, 256, 1024);
    }
    const h16* pt = &plds[buf][0];
    const h16* gt = &glds[buf][0];

    // --- S tiles from LDS: lane = column m, contiguous K=i ---
    v8f s0 = zero, s1 = zero;
#pragma unroll
    for (int k = 0; k < II / 32; ++k) {
      s0 = wmma_f16(af[k],
                    *(const v16h*)(pt + (size_t)col * II + k * 32 + hf * 16),
                    s0);
      s1 = wmma_f16(
          af[k], *(const v16h*)(pt + (size_t)(16 + col) * II + k * 32 + hf * 16),
          s1);
    }

    // --- Online softmax: lane holds rows v + 8*hf within its 16-lane half ---
    float alpha[8];
#pragma unroll
    for (int v = 0; v < 8; ++v) {
      float a0 = s0[v] * SCALE, a1 = s1[v] * SCALE;
      float mx = fmaxf(a0, a1);
#pragma unroll
      for (int d = 1; d < 16; d <<= 1) mx = fmaxf(mx, __shfl_xor(mx, d, 32));
      float mnew = fmaxf(mrow[v], mx);
      float al = __expf(mrow[v] - mnew);
      float p0 = __expf(a0 - mnew), p1 = __expf(a1 - mnew);
      float rs = p0 + p1;
#pragma unroll
      for (int d = 1; d < 16; d <<= 1) rs += __shfl_xor(rs, d, 32);
      lrow[v] = lrow[v] * al + rs;
      mrow[v] = mnew;
      alpha[v] = al;
      pstage[wave][(v + 8 * hf) * 32 + col] = (h16)p0;
      pstage[wave][(v + 8 * hf) * 32 + 16 + col] = (h16)p1;
    }
    asm volatile("s_wait_dscnt 0" ::: "memory");  // wave-local DS RAW fence
    v16h pf = load_a_frag(&pstage[wave][col * 32], hf);

    // --- O += P @ g^T from LDS: lane = column i, contiguous K=m ---
#pragma unroll
    for (int t = 0; t < 16; ++t) {
      v16h gf = *(const v16h*)(gt + (size_t)(t * 16 + col) * 32 + hf * 16);
      v8f cin;
#pragma unroll
      for (int v = 0; v < 8; ++v) cin[v] = acc[t][v] * alpha[v];
      acc[t] = wmma_f16(pf, gf, cin);
    }
    __syncthreads();  // everyone done with tiles[buf] before it is re-filled
    buf ^= 1;
  }

  // Normalize and emit oT[n][i]; lane holds column i, rows n0 + v + 8*hf.
#pragma unroll
  for (int t = 0; t < 16; ++t) {
#pragma unroll
    for (int v = 0; v < 8; ++v) {
      float o = acc[t][v] / lrow[v];
      oT[((size_t)b * NN + n0 + v + 8 * hf) * II + t * 16 + col] = (h16)o;
    }
  }
}

// ---------------------------------------------------------------------------
// Phase 3: out = out_w (C,I) @ O (I,N) + out_b + x, f32 result.
// ---------------------------------------------------------------------------
__global__ void __launch_bounds__(128) outproj_kernel(
    const h16* __restrict__ wout, const h16* __restrict__ oT,
    const float* __restrict__ outb, const float* __restrict__ x,
    float* __restrict__ out) {
  const int lane = threadIdx.x & 31, wave = threadIdx.x >> 5;
  const int col = lane & 15, hf = lane >> 4;
  const int n0 = (blockIdx.x * 4 + wave) * 16;
  const int c0 = blockIdx.y * 16;
  const int b = blockIdx.z;

  const h16* arow = wout + (size_t)(c0 + col) * II;
  const h16* brow = oT + ((size_t)b * NN + n0 + col) * II + hf * 16;
  v8f acc = {};
#pragma unroll
  for (int k = 0; k < II / 32; ++k)
    acc = wmma_f16(load_a_frag(arow + k * 32, hf),
                   *(const v16h*)(brow + k * 32), acc);

  v8f bo = *(const v8f*)(outb + c0 + hf * 8);
#pragma unroll
  for (int v = 0; v < 8; ++v) {
    size_t idx = ((size_t)b * CC + c0 + hf * 8 + v) * NN + n0 + col;
    out[idx] = acc[v] + bo[v] + x[idx];
  }
}

// ---------------------------------------------------------------------------
extern "C" void kernel_launch(void* const* d_in, const int* in_sizes, int n_in,
                              void* d_out, int out_size, void* d_ws,
                              size_t ws_size, hipStream_t stream) {
  const float* x = (const float*)d_in[0];
  const float* thw = (const float*)d_in[1];
  const float* thb = (const float*)d_in[2];
  const float* phw = (const float*)d_in[3];
  const float* phb = (const float*)d_in[4];
  const float* gw = (const float*)d_in[5];
  const float* gb = (const float*)d_in[6];
  const float* outw = (const float*)d_in[7];
  const float* outb = (const float*)d_in[8];
  float* out = (float*)d_out;

  char* ws = (char*)d_ws;
  size_t off = 0;
  auto alloc = [&](size_t bytes) -> char* {
    char* p = ws + off;
    off = (off + bytes + 255) & ~(size_t)255;
    return p;
  };
  h16* xt = (h16*)alloc((size_t)BB * NN * CC * 2);  // 16 MB
  h16* wth = (h16*)alloc((size_t)II * CC * 2);
  h16* wph = (h16*)alloc((size_t)II * CC * 2);
  h16* wg = (h16*)alloc((size_t)II * CC * 2);
  h16* wout = (h16*)alloc((size_t)CC * II * 2);
  h16* thetaT = (h16*)alloc((size_t)BB * NN * II * 2);  // 8 MB
  h16* phiT = (h16*)alloc((size_t)BB * NN * II * 2);
  h16* gmat = (h16*)alloc((size_t)BB * II * NN * 2);
  h16* oT = (h16*)alloc((size_t)BB * NN * II * 2);

  cvt_f16_kernel<<<(II * CC + 255) / 256, 256, 0, stream>>>(thw, wth, II * CC);
  cvt_f16_kernel<<<(II * CC + 255) / 256, 256, 0, stream>>>(phw, wph, II * CC);
  cvt_f16_kernel<<<(II * CC + 255) / 256, 256, 0, stream>>>(gw, wg, II * CC);
  cvt_f16_kernel<<<(CC * II + 255) / 256, 256, 0, stream>>>(outw, wout,
                                                            CC * II);
  transpose_x_kernel<<<(BB * NN * CC) / 256, 256, 0, stream>>>(x, xt);

  proj_kernel<<<dim3(NN / 64, II / 16, BB), 128, 0, stream>>>(
      xt, wth, wph, wg, thb, phb, gb, thetaT, phiT, gmat);
  attn_kernel<<<dim3(NN / 64, 1, BB), 128, 0, stream>>>(thetaT, phiT, gmat, oT);
  outproj_kernel<<<dim3(NN / 64, CC / 16, BB), 128, 0, stream>>>(wout, oT, outb,
                                                                 x, out);
}
